// Char_LSTM_83519934038527
// MI455X (gfx1250) — matrix operand order repaired and checked by reading, not compiled, not run on hardware
//
#include <hip/hip_runtime.h>

// ---------------------------------------------------------------------------
// Char-LSTM + causal attention forward for MI455X (gfx1250, wave32, WMMA).
// All GEMMs run on v_wmma_f32_16x16x32_bf16. LSTM is a persistent per-layer
// kernel with h/c state in LDS and TDM (tensor_load_to_lds) staging of x_t;
// attention stages Q and V stripes through LDS and keeps P@V operands local.
// ---------------------------------------------------------------------------

typedef __attribute__((ext_vector_type(16))) __bf16 v16bf;
typedef __attribute__((ext_vector_type(8)))  float  v8f;
typedef unsigned short u16;

// TDM descriptor vector types (clang-23 6-arg builtin form)
typedef __attribute__((ext_vector_type(4))) unsigned tdm_v4u;
typedef __attribute__((ext_vector_type(8))) int      tdm_v8i;
typedef __attribute__((ext_vector_type(4))) int      tdm_v4i;

#if defined(__has_builtin)
#if __has_builtin(__builtin_amdgcn_tensor_load_to_lds) && \
    __has_builtin(__builtin_amdgcn_s_wait_tensorcnt)
#define HAVE_TDM 1
#endif
#endif
#ifndef HAVE_TDM
#define HAVE_TDM 0
#endif

#define B_     32
#define L_     1024
#define EMB_   128
#define H_     384
#define NL_    4
#define VOCAB_ 100
#define CTXIN_ 300
#define G4H_   1536   // 4*H

// ----------------------------- scalar helpers ------------------------------

__device__ __forceinline__ u16 f2bf_bits(float f) {
  union { float f; unsigned u; } x; x.f = f;
  unsigned u = x.u;
  unsigned r = u + 0x7FFFu + ((u >> 16) & 1u);   // round-to-nearest-even
  return (u16)(r >> 16);
}

__device__ __forceinline__ __bf16 bits_to_bf(u16 v) {
  union { u16 u; __bf16 b; } cv; cv.u = v; return cv.b;
}

__device__ __forceinline__ float sigmoidf(float x) {
  return 1.0f / (1.0f + __expf(-x));
}

__device__ __forceinline__ v8f vzero8() {
  v8f z = {0.f, 0.f, 0.f, 0.f, 0.f, 0.f, 0.f, 0.f};
  return z;
}

// ------------------------- WMMA tile load helpers --------------------------
// A (16x32 bf16, MxK): lanes 0-15 row m, K {0..15}; lanes 16-31 row m,
// K {16..31}.
__device__ __forceinline__ v16bf load_a_rm(const u16* X, int ld, int m0,
                                           int kb, int kmax, int lane) {
  const int m  = m0 + (lane & 15);
  const int ko = (lane >> 4) << 4;
  const u16* row = X + (size_t)m * ld + kb + ko;
  v16bf a;
#pragma unroll
  for (int i = 0; i < 16; ++i) {
    int k0 = kb + ko + i;
    a[i] = bits_to_bf(k0 < kmax ? row[i] : (u16)0);
  }
  return a;
}

// B (32x16 bf16, KxN) built from row-major W[N][K]  (computes X @ W^T).
__device__ __forceinline__ v16bf load_b_nmajor(const u16* W, int ld, int n0,
                                               int kb, int kmax, int nmax,
                                               int lane) {
  const int n  = n0 + (lane & 15);
  const int ko = (lane >> 4) << 4;
  v16bf b;
  if (n < nmax) {
    const u16* row = W + (size_t)n * ld + kb + ko;
#pragma unroll
    for (int i = 0; i < 16; ++i) {
      int k = kb + ko + i;
      b[i] = bits_to_bf(k < kmax ? row[i] : (u16)0);
    }
  } else {
#pragma unroll
    for (int i = 0; i < 16; ++i) b[i] = bits_to_bf((u16)0);
  }
  return b;
}

// B built from row-major V[K][N] (strided per lane; used on LDS-staged data).
__device__ __forceinline__ v16bf load_b_kmajor(const u16* V, int ld, int n0,
                                               int kb, int kmax, int lane) {
  const int n  = n0 + (lane & 15);
  const int ko = (lane >> 4) << 4;
  v16bf b;
#pragma unroll
  for (int i = 0; i < 16; ++i) {
    int k = kb + ko + i;
    b[i] = bits_to_bf(k < kmax ? V[(size_t)k * ld + n] : (u16)0);
  }
  return b;
}

__device__ __forceinline__ v8f wmma_bf(v16bf a, v16bf b, v8f c) {
  return __builtin_amdgcn_wmma_f32_16x16x32_bf16(
      false, a, false, b, (short)0, c, false, false);
}

// ------------------------------ small kernels ------------------------------

__global__ void __launch_bounds__(256)
f32_to_bf16_kernel(const float* __restrict__ src, u16* __restrict__ dst,
                   size_t n) {
  size_t i = (size_t)blockIdx.x * blockDim.x + threadIdx.x;
  if (i < n) dst[i] = f2bf_bits(src[i]);
}

__global__ void __launch_bounds__(256)
embed_kernel(const int* __restrict__ x, const float* __restrict__ emb,
             u16* __restrict__ inp0, size_t M) {
  size_t i = (size_t)blockIdx.x * blockDim.x + threadIdx.x;
  if (i < M * EMB_) {
    size_t m = i >> 7;
    int c = (int)(i & 127);
    int tok = x[m];
    inp0[m * (2 * EMB_) + c] = f2bf_bits(emb[(size_t)tok * EMB_ + c]);
  }
}

// --------------------- generic GEMM:  Y = act(X @ W^T + b) -----------------
template <int OUT_BF16, int ACT_TANH>
__global__ void __launch_bounds__(256)
gemm_xwt(const u16* __restrict__ X, int ldx, const u16* __restrict__ W,
         int ldw, const float* __restrict__ bias, void* __restrict__ Y,
         int ldy, int M, int N, int Npad, int K) {
  size_t gw = ((size_t)blockIdx.x * blockDim.x + threadIdx.x) >> 5;
  int lane = threadIdx.x & 31;
  int ntiles = Npad >> 4;
  int mt = (int)(gw / ntiles);
  int nt = (int)(gw % ntiles);
  if (mt * 16 >= M) return;

  v8f acc = vzero8();
  for (int kb = 0; kb < K; kb += 32) {
    __builtin_prefetch(W + (size_t)(nt * 16 + (lane & 15)) * ldw + kb + 64, 0, 1);
    v16bf a = load_a_rm(X, ldx, mt * 16, kb, K, lane);
    v16bf b = load_b_nmajor(W, ldw, nt * 16, kb, K, N, lane);
    acc = wmma_bf(a, b, acc);
  }

  int n  = nt * 16 + (lane & 15);
  int r0 = (lane >> 4) * 8;
  float bv = (bias != nullptr && n < N) ? bias[n] : 0.f;
#pragma unroll
  for (int r = 0; r < 8; ++r) {
    int m = mt * 16 + r0 + r;
    if (n < N && m < M) {
      float v = acc[r] + bv;
      if (ACT_TANH) v = tanhf(v);
      if (OUT_BF16) ((u16*)Y)[(size_t)m * ldy + n] = f2bf_bits(v);
      else          ((float*)Y)[(size_t)m * ldy + n] = v;
    }
  }
}

// ------------- combine: Y = tanh(concat(X1,X2) @ W^T + b), bf16 ------------
__global__ void __launch_bounds__(256)
gemm_concat_tanh(const u16* __restrict__ X1, const u16* __restrict__ X2,
                 const u16* __restrict__ W, const float* __restrict__ bias,
                 u16* __restrict__ Y, int M) {
  size_t gw = ((size_t)blockIdx.x * blockDim.x + threadIdx.x) >> 5;
  int lane = threadIdx.x & 31;
  const int ntiles = H_ / 16;  // 24
  int mt = (int)(gw / ntiles);
  int nt = (int)(gw % ntiles);
  if (mt * 16 >= M) return;

  v8f acc = vzero8();
  for (int kb = 0; kb < H_; kb += 32) {
    v16bf a = load_a_rm(X1, H_, mt * 16, kb, H_, lane);
    acc = wmma_bf(a, load_b_nmajor(W, 2 * H_, nt * 16, kb, 2 * H_, H_, lane), acc);
  }
  for (int kb = 0; kb < H_; kb += 32) {
    v16bf a = load_a_rm(X2, H_, mt * 16, kb, H_, lane);
    acc = wmma_bf(a, load_b_nmajor(W, 2 * H_, nt * 16, H_ + kb, 2 * H_, H_, lane), acc);
  }
  int n  = nt * 16 + (lane & 15);
  int r0 = (lane >> 4) * 8;
  float bv = bias[n];
#pragma unroll
  for (int r = 0; r < 8; ++r)
    Y[(size_t)(mt * 16 + r0 + r) * H_ + n] = f2bf_bits(tanhf(acc[r] + bv));
}

// ------------------------------- LSTM layer --------------------------------
// Persistent single workgroup (24 waves). h/c state + x_t staged in LDS; x_t
// is DMA'd by the Tensor Data Mover when available (TENSORcnt-tracked).
__global__ void __launch_bounds__(768)
lstm_layer_kernel(const u16* __restrict__ Xseq, int Din,
                  const u16* __restrict__ Wih,   // [1536, Din] bf16
                  const u16* __restrict__ Whh,   // [1536, 384] bf16
                  const float* __restrict__ bih, const float* __restrict__ bhh,
                  const float* __restrict__ h0in, const float* __restrict__ c0in,
                  u16* __restrict__ Yseq,        // [B, L, H] bf16
                  float* __restrict__ hf_out, float* __restrict__ cf_out) {
  extern __shared__ char smem[];
  float* c_s  = (float*)smem;                 // [32][384] f32
  float* hf_s = c_s + B_ * H_;                // [32][384] f32
  u16* hA  = (u16*)(hf_s + B_ * H_);          // [32][384] bf16 ping
  u16* hB  = hA + B_ * H_;                    // pong
  u16* x_s = hB + B_ * H_;                    // [32][Din] bf16

  const int tid  = threadIdx.x;
  const int wave = tid >> 5;
  const int lane = tid & 31;
  const int ln   = lane & 15;
  const int lh   = lane >> 4;
  const int nwv  = blockDim.x >> 5;           // 24

  for (int i = tid; i < B_ * H_; i += blockDim.x) {
    float hv = h0in[i];
    c_s[i]  = c0in[i];
    hf_s[i] = hv;
    hA[i]   = f2bf_bits(hv);
  }
  __syncthreads();

#if HAVE_TDM
  const unsigned lds_off = (unsigned)((const char*)x_s - (const char*)smem);
#endif

  for (int t = 0; t < L_; ++t) {
    // ---- stage x_t[32][Din] into LDS ----
#if HAVE_TDM
    if (wave == 0) {
      // 2D tile: 32 rows of Din bf16 elements, row stride L_*Din elements.
      unsigned long long ga =
          (unsigned long long)(__UINTPTR_TYPE__)Xseq +
          (unsigned long long)t * ((unsigned)Din * 2u);
      tdm_v4u g0;
      g0[0] = 1u;                                       // count=1, user mode
      g0[1] = lds_off;                                  // lds_addr
      g0[2] = (unsigned)ga;                             // global_addr[31:0]
      g0[3] = (unsigned)((ga >> 32) & 0x01FFFFFFu) | (2u << 30);  // type=2
      tdm_v8i g1;
      g1[0] = (int)(1u << 16);                          // data_size=1 (2B)
      g1[1] = (int)(((unsigned)Din & 0xFFFFu) << 16);   // tensor_dim0 lo
      g1[2] = (int)(((unsigned)Din >> 16) | (32u << 16)); // td0 hi | td1=32
      g1[3] = (int)((unsigned)Din << 16);               // tile_dim0 = Din
      g1[4] = 32;                                       // tile_dim1=32, td2=0
      g1[5] = (int)((unsigned)(L_ * Din));              // dim0_stride lo32
      g1[6] = 0;
      g1[7] = 0;
      tdm_v4i z4 = {0, 0, 0, 0};
      tdm_v8i z8 = {0, 0, 0, 0, 0, 0, 0, 0};
      __builtin_amdgcn_tensor_load_to_lds(g0, g1, z4, z4, z8, 0);
      __builtin_amdgcn_s_wait_tensorcnt(0);
    }
#else
    {
      const int xn = B_ * Din;
      for (int i = tid; i < xn; i += blockDim.x) {
        int b = i / Din, k = i - b * Din;
        x_s[i] = Xseq[((size_t)b * L_ + t) * Din + k];
      }
    }
#endif
    __syncthreads();

    const u16* hprev = (t & 1) ? hB : hA;
    u16*       hnext = (t & 1) ? hA : hB;

    for (int p = wave; p < 48; p += nwv) {     // 48 = 2 m-tiles * 24 n-blocks
      const int mt = p & 1;
      const int nb = p >> 1;
      v8f a0 = vzero8(), a1 = vzero8(), a2 = vzero8(), a3 = vzero8();

      for (int kb = 0; kb < Din; kb += 32) {
        v16bf a = load_a_rm(x_s, Din, mt * 16, kb, Din, lane);
        __builtin_prefetch(Wih + (size_t)(nb * 16 + ln) * Din + kb + 64, 0, 1);
        a0 = wmma_bf(a, load_b_nmajor(Wih, Din, 0 * H_ + nb * 16, kb, Din, G4H_, lane), a0);
        a1 = wmma_bf(a, load_b_nmajor(Wih, Din, 1 * H_ + nb * 16, kb, Din, G4H_, lane), a1);
        a2 = wmma_bf(a, load_b_nmajor(Wih, Din, 2 * H_ + nb * 16, kb, Din, G4H_, lane), a2);
        a3 = wmma_bf(a, load_b_nmajor(Wih, Din, 3 * H_ + nb * 16, kb, Din, G4H_, lane), a3);
      }
      for (int kb = 0; kb < H_; kb += 32) {
        v16bf a = load_a_rm(hprev, H_, mt * 16, kb, H_, lane);
        a0 = wmma_bf(a, load_b_nmajor(Whh, H_, 0 * H_ + nb * 16, kb, H_, G4H_, lane), a0);
        a1 = wmma_bf(a, load_b_nmajor(Whh, H_, 1 * H_ + nb * 16, kb, H_, G4H_, lane), a1);
        a2 = wmma_bf(a, load_b_nmajor(Whh, H_, 2 * H_ + nb * 16, kb, H_, G4H_, lane), a2);
        a3 = wmma_bf(a, load_b_nmajor(Whh, H_, 3 * H_ + nb * 16, kb, H_, G4H_, lane), a3);
      }

      const int j = nb * 16 + ln;
      const float b0 = bih[0 * H_ + j] + bhh[0 * H_ + j];
      const float b1 = bih[1 * H_ + j] + bhh[1 * H_ + j];
      const float b2 = bih[2 * H_ + j] + bhh[2 * H_ + j];
      const float b3 = bih[3 * H_ + j] + bhh[3 * H_ + j];
#pragma unroll
      for (int r = 0; r < 8; ++r) {
        const int m   = mt * 16 + lh * 8 + r;
        const int idx = m * H_ + j;
        float ig = sigmoidf(a0[r] + b0);
        float fg = sigmoidf(a1[r] + b1);
        float gg = tanhf   (a2[r] + b2);
        float og = sigmoidf(a3[r] + b3);
        float cn = fg * c_s[idx] + ig * gg;
        float hn = og * tanhf(cn);
        c_s[idx]  = cn;
        hf_s[idx] = hn;
        u16 hb = f2bf_bits(hn);
        hnext[idx] = hb;
        Yseq[((size_t)m * L_ + t) * H_ + j] = hb;
      }
    }
    __syncthreads();
  }

  for (int i = tid; i < B_ * H_; i += blockDim.x) {
    hf_out[i] = hf_s[i];
    cf_out[i] = c_s[i];
  }
}

// ----------------------- causal attention (per q-tile) ---------------------
// block = one (batch, 16-query tile). Q staged in LDS; WMMA QK^T -> LDS
// scores -> softmax -> bf16 P in LDS; P@V with V stripes staged in LDS
// (aliasing the dead score buffer), 3 accumulator tiles per wave.
__global__ void __launch_bounds__(256)
attn_kernel(const u16* __restrict__ Out, const u16* __restrict__ Keys,
            u16* __restrict__ Ctx) {
  extern __shared__ char smem[];
  float* S  = (float*)smem;                            // [16][1024] 64KB
  u16*   P  = (u16*)(smem + 16 * L_ * 4);              // [16][1024] 32KB
  u16*   Qs = (u16*)(smem + 16 * L_ * 4 + 16 * L_ * 2);// [16][384]  12KB
  u16*   Vs = (u16*)smem;                              // [32][384] alias of S

  const int bq = blockIdx.x;
  const int b  = bq >> 6;
  const int q0 = (bq & 63) << 4;
  const u16* Q  = Out  + ((size_t)b * L_ + q0) * H_;
  const u16* Kb = Keys + (size_t)b * L_ * H_;
  const u16* V  = Out  + (size_t)b * L_ * H_;
  const int nk = q0 + 16;                  // valid keys (causal)
  const int ktiles = (q0 >> 4) + 1;
  const int tid = threadIdx.x;
  const int wave = tid >> 5, lane = tid & 31;
  const int nwv = blockDim.x >> 5;         // 8
  const float scale = 0.051031036307982884f;  // 1/sqrt(384)

  // stage Q tile (16 x 384, contiguous rows) into LDS
  for (int ci = tid; ci < (16 * H_) / 8; ci += blockDim.x)
    *(uint4*)(Qs + ci * 8) = *(const uint4*)(Q + ci * 8);
  __syncthreads();

  // ---- scores = Q K^T (causal-pruned key tiles) ----
  for (int kt = wave; kt < ktiles; kt += nwv) {
    v8f acc = vzero8();
    for (int kb = 0; kb < H_; kb += 32) {
      v16bf a  = load_a_rm(Qs, H_, 0, kb, H_, lane);
      v16bf bb = load_b_nmajor(Kb, H_, kt * 16, kb, H_, L_, lane);
      acc = wmma_bf(a, bb, acc);
    }
    int n  = kt * 16 + (lane & 15);
    int r0 = (lane >> 4) * 8;
#pragma unroll
    for (int r = 0; r < 8; ++r) {
      int row = r0 + r;
      float v = acc[r] * scale;
      if (n > q0 + row) v = -1e30f;        // causal mask
      S[row * L_ + n] = v;
    }
  }
  __syncthreads();

  // ---- row softmax over [0, nk) ----
  for (int row = wave; row < 16; row += nwv) {
    float mx = -1e30f;
    for (int j = lane; j < nk; j += 32) mx = fmaxf(mx, S[row * L_ + j]);
    for (int o = 16; o > 0; o >>= 1) mx = fmaxf(mx, __shfl_xor(mx, o, 32));
    float sum = 0.f;
    for (int j = lane; j < nk; j += 32) {
      float e = __expf(S[row * L_ + j] - mx);
      S[row * L_ + j] = e;
      sum += e;
    }
    for (int o = 16; o > 0; o >>= 1) sum += __shfl_xor(sum, o, 32);
    float inv = 1.f / sum;
    for (int j = lane; j < L_; j += 32)
      P[row * L_ + j] = f2bf_bits(j < nk ? S[row * L_ + j] * inv : 0.f);
  }
  __syncthreads();   // S is dead past here; Vs may overwrite it

  // ---- ctx = P @ V, V stripes staged through LDS ----
  v8f acc[3] = {vzero8(), vzero8(), vzero8()};
  for (int kb = 0; kb < nk; kb += 32) {
    // coalesced stage of V rows [kb, kb+32) x 384 (rows >= nk multiply 0-P)
    for (int ci = tid; ci < (32 * H_) / 8; ci += blockDim.x) {
      int e = ci * 8;
      int k = e / H_, c = e - k * H_;
      *(uint4*)(Vs + e) = *(const uint4*)(V + (size_t)(kb + k) * H_ + c);
    }
    __syncthreads();
    v16bf a = load_a_rm(P, L_, 0, kb, L_, lane);
#pragma unroll
    for (int j = 0; j < 3; ++j) {
      v16bf bb = load_b_kmajor(Vs, H_, (wave * 3 + j) * 16, 0, 32, lane);
      acc[j] = wmma_bf(a, bb, acc[j]);
    }
    __syncthreads();
  }

  u16* Crow = Ctx + ((size_t)b * L_ + q0) * H_;
  const int r0 = (lane >> 4) * 8;
#pragma unroll
  for (int j = 0; j < 3; ++j) {
    int n = (wave * 3 + j) * 16 + (lane & 15);
#pragma unroll
    for (int r = 0; r < 8; ++r)
      Crow[(size_t)(r0 + r) * H_ + n] = f2bf_bits(acc[j][r]);
  }
}

// --------------------------------- driver ----------------------------------

extern "C" void kernel_launch(void* const* d_in, const int* in_sizes, int n_in,
                              void* d_out, int out_size, void* d_ws,
                              size_t ws_size, hipStream_t stream) {
  (void)in_sizes; (void)n_in; (void)out_size; (void)ws_size;

  const int*   x     = (const int*)  d_in[0];
  const float* ctx   = (const float*)d_in[1];
  const float* h0    = (const float*)d_in[2];
  const float* c0    = (const float*)d_in[3];
  const float* emb   = (const float*)d_in[4];
  const float* ctx_W = (const float*)d_in[5];
  const float* ctx_b = (const float*)d_in[6];
  const float* Wih0  = (const float*)d_in[7];
  const float* WihR  = (const float*)d_in[8];
  const float* Whh   = (const float*)d_in[9];
  const float* bih   = (const float*)d_in[10];
  const float* bhh   = (const float*)d_in[11];
  const float* Wa    = (const float*)d_in[12];
  const float* combW = (const float*)d_in[13];
  const float* combB = (const float*)d_in[14];
  const float* fcW   = (const float*)d_in[15];
  const float* fcB   = (const float*)d_in[16];

  const size_t M = (size_t)B_ * L_;  // 32768 rows

  // ---- workspace carve-up (bf16 regions, 256B aligned) ----
  char* ws = (char*)d_ws;
  size_t off = 0;
  auto carve = [&](size_t elems) -> u16* {
    u16* p = (u16*)(ws + off);
    off = (off + elems * 2 + 255) & ~(size_t)255;
    return p;
  };
  u16* wih0_b  = carve((size_t)G4H_ * 2 * EMB_);   // [1536,256]
  u16* wihr_b  = carve((size_t)3 * G4H_ * H_);     // [3,1536,384]
  u16* whh_b   = carve((size_t)NL_ * G4H_ * H_);   // [4,1536,384]
  u16* wa_b    = carve((size_t)H_ * H_);
  u16* combw_b = carve((size_t)H_ * 2 * H_);
  u16* fcw_b   = carve((size_t)VOCAB_ * H_);
  u16* ctxw_b  = carve((size_t)EMB_ * CTXIN_);
  u16* inp0    = carve(M * 2 * EMB_);              // [M,256]
  u16* bufP    = carve(M * H_);                    // ping
  u16* bufQ    = carve(M * H_);                    // pong
  u16* ctxv    = carve(M * H_);                    // attention output
  // time-disjoint aliases of bufP:
  u16* ctxbf = bufP;   // bf16 context  (before layer 0 writes bufP)
  u16* keys  = bufP;   // keys          (after layer 3, bufP dead)
  u16* comb  = bufP;   // combined      (after attention, keys dead)

  auto cvt = [&](const float* s, u16* d, size_t n) {
    f32_to_bf16_kernel<<<(unsigned)((n + 255) / 256), 256, 0, stream>>>(s, d, n);
  };
  cvt(Wih0, wih0_b, (size_t)G4H_ * 2 * EMB_);
  cvt(WihR, wihr_b, (size_t)3 * G4H_ * H_);
  cvt(Whh,  whh_b,  (size_t)NL_ * G4H_ * H_);
  cvt(Wa,   wa_b,   (size_t)H_ * H_);
  cvt(combW, combw_b, (size_t)H_ * 2 * H_);
  cvt(fcW,  fcw_b,  (size_t)VOCAB_ * H_);
  cvt(ctx_W, ctxw_b, (size_t)EMB_ * CTXIN_);
  cvt(ctx,  ctxbf,  M * CTXIN_);

  // embedding -> inp0[:, :128]
  embed_kernel<<<(unsigned)((M * EMB_ + 255) / 256), 256, 0, stream>>>(
      x, emb, inp0, M);

  // ctx projection -> inp0[:, 128:256]   (M x 128, K=300 guarded)
  {
    size_t waves = (M / 16) * (size_t)(EMB_ / 16);
    gemm_xwt<1, 0><<<(unsigned)((waves + 7) / 8), 256, 0, stream>>>(
        ctxbf, CTXIN_, ctxw_b, CTXIN_, ctx_b, (void*)(inp0 + EMB_),
        2 * EMB_, (int)M, EMB_, EMB_, CTXIN_);
  }

  float* logits = (float*)d_out;                       // [M,100]
  float* hs = logits + M * VOCAB_;                     // [4,32,384]
  float* cs = hs + (size_t)NL_ * B_ * H_;              // [4,32,384]

  // ---- 4 LSTM layers, persistent kernel each ----
  const size_t lstm_smem = (size_t)B_ * H_ * 4 * 2     // c_s + hf_s (f32)
                         + (size_t)B_ * H_ * 2 * 2     // hA + hB (bf16)
                         + (size_t)B_ * H_ * 2;        // x_s (bf16, Din<=384)
  const u16* lin[4]  = {inp0, bufP, bufQ, bufP};
  u16*       lout[4] = {bufP, bufQ, bufP, bufQ};
  for (int l = 0; l < NL_; ++l) {
    int Din = (l == 0) ? 2 * EMB_ : H_;
    const u16* Wih_l = (l == 0) ? wih0_b : (wihr_b + (size_t)(l - 1) * G4H_ * H_);
    lstm_layer_kernel<<<1, 768, (unsigned)lstm_smem, stream>>>(
        lin[l], Din, Wih_l, whh_b + (size_t)l * G4H_ * H_,
        bih + (size_t)l * G4H_, bhh + (size_t)l * G4H_,
        h0 + (size_t)l * B_ * H_, c0 + (size_t)l * B_ * H_,
        lout[l], hs + (size_t)l * B_ * H_, cs + (size_t)l * B_ * H_);
  }
  u16* outseq = bufQ;  // final layer output [B,L,H] bf16

  // keys = out @ Wa^T
  {
    size_t waves = (M / 16) * (size_t)(H_ / 16);
    gemm_xwt<1, 0><<<(unsigned)((waves + 7) / 8), 256, 0, stream>>>(
        outseq, H_, wa_b, H_, nullptr, (void*)keys, H_, (int)M, H_, H_, H_);
  }

  // causal attention -> ctxv   (LDS: 64KB scores/V + 32KB P + 12KB Q)
  attn_kernel<<<B_ * (L_ / 16), 256,
                16 * L_ * 4 + 16 * L_ * 2 + 16 * H_ * 2, stream>>>(
      outseq, keys, ctxv);

  // combined = tanh([out||ctx] @ comb_W^T + b)
  {
    size_t waves = (M / 16) * (size_t)(H_ / 16);
    gemm_concat_tanh<<<(unsigned)((waves + 7) / 8), 256, 0, stream>>>(
        outseq, ctxv, combw_b, combB, comb, (int)M);
  }

  // logits = combined @ fc_W^T + fc_b   (N=100 padded to 112 tiles)
  {
    size_t waves = (M / 16) * (size_t)(112 / 16);
    gemm_xwt<0, 0><<<(unsigned)((waves + 7) / 8), 256, 0, stream>>>(
        comb, H_, fcw_b, H_, fcB, (void*)logits, VOCAB_, (int)M, VOCAB_, 112, H_);
  }
}
